// SRSChannelEstimator_69999376990673
// MI455X (gfx1250) — matrix operand order repaired and checked by reading, not compiled, not run on hardware
//
#include <hip/hip_runtime.h>
#include <hip/hip_bf16.h>

// ---------------------------------------------------------------------------
// SRS channel estimator for MI455X (gfx1250, wave32, WMMA).
// K1: IFFT-as-DFT power via v_wmma_f32_16x16x32_f16, 48 t-cols/block,
//     integer-phase twiddles gathered from a 1632-entry LDS phasor table.
// K2: windowed argmax -> timing offsets
// K3: fused derotate/avg/interp/residual + 48x48 complex MMSE GEMM via WMMA,
//     all intermediates + phasor table resident in the WGP's 320KB LDS.
// ---------------------------------------------------------------------------

typedef _Float16 f16_t;
typedef __attribute__((ext_vector_type(16))) _Float16 v16h;
typedef __attribute__((ext_vector_type(8)))  _Float16 v8h;
typedef __attribute__((ext_vector_type(8)))  float    v8f;

#define L_SEQ   1632
#define NRX     64
#define NPORT   16
#define NCHUNK  34          // 1632 / 48
#define NAVG    408         // 1632 / 4
#define TWO_PI_OVER_L (6.283185307179586f / 1632.0f)

__device__ __forceinline__ v16h cat8(v8h lo, v8h hi) {
  return __builtin_shufflevector(lo, hi, 0,1,2,3,4,5,6,7,8,9,10,11,12,13,14,15);
}

__device__ __forceinline__ v8f wmma_f16(v16h a, v16h b, v8f c) {
  // D = A(16x32 f16) * B(32x16 f16) + C(16x16 f32)
  return __builtin_amdgcn_wmma_f32_16x16x32_f16(false, a, false, b,
                                                (short)0, c, false, false);
}

// pack 8+8 contiguous floats into a v16h per the f16 A-matrix lane layout
__device__ __forceinline__ v16h pack16(const float* p0, const float* p1) {
  float4 a0 = *(const float4*)p0;
  float4 a1 = *(const float4*)(p0 + 4);
  float4 b0 = *(const float4*)p1;
  float4 b1 = *(const float4*)(p1 + 4);
  v16h v;
  v[0]=(f16_t)a0.x; v[1]=(f16_t)a0.y; v[2]=(f16_t)a0.z; v[3]=(f16_t)a0.w;
  v[4]=(f16_t)a1.x; v[5]=(f16_t)a1.y; v[6]=(f16_t)a1.z; v[7]=(f16_t)a1.w;
  v[8]=(f16_t)b0.x; v[9]=(f16_t)b0.y; v[10]=(f16_t)b0.z; v[11]=(f16_t)b0.w;
  v[12]=(f16_t)b1.x; v[13]=(f16_t)b1.y; v[14]=(f16_t)b1.z; v[15]=(f16_t)b1.w;
  return v;
}

// ---------------------------------------------------------------------------
// K1: h_power[b][t] = sum_r | sum_n ls[b,r,n] e^{+2pi i n t / L} |^2
// grid (34 t-tiles of 48, 8 batches), 128 threads (4 waves = 16-row r-blocks)
// ---------------------------------------------------------------------------
__global__ __launch_bounds__(128)
void k1_dft_power(const float* __restrict__ ls_re,
                  const float* __restrict__ ls_im,
                  float* __restrict__ h_power) {
  __shared__ float2 tab[L_SEQ];          // e^{i 2pi q / L}
  __shared__ f16_t  cosB[48 * 32];
  __shared__ f16_t  sinB[48 * 32];
  __shared__ float  pow_s[48];

  const int tid   = threadIdx.x;
  const int wave  = tid >> 5;
  const int lane  = tid & 31;
  const int tt    = blockIdx.x * 48;     // t-tile base (3 N-tiles of 16)
  const int b     = blockIdx.y;
  const int mrow  = lane & 15;           // A-layout: M = lane%16
  const int khalf = lane >> 4;           // A-layout: K-half select
  const int r     = wave * 16 + mrow;

  // phasor table: the only transcendentals in this kernel
  for (int q = tid; q < L_SEQ; q += 128) {
    float s, c; __sincosf(TWO_PI_OVER_L * (float)q, &s, &c);
    tab[q] = make_float2(c, s);
  }
  if (tid < 48) pow_s[tid] = 0.f;

  const float* rowre = ls_re + ((size_t)(b * NRX + r)) * L_SEQ;
  const float* rowim = ls_im + ((size_t)(b * NRX + r)) * L_SEQ;

  v8f accRe[3] = {}, accIm[3] = {};

  for (int k0 = 0; k0 < L_SEQ; k0 += 32) {
    __syncthreads();
    // twiddle staging: B[k][col] = e^{+i 2pi (k0+k)(tt+col)/L} via table gather
    for (int i = tid; i < 48 * 32; i += 128) {
      int col = i >> 5, k = i & 31;
      int q = ((k0 + k) * (tt + col)) % L_SEQ;   // exact integer phase
      float2 e = tab[q];
      cosB[i] = (f16_t)e.x;
      sinB[i] = (f16_t)e.y;
    }
    __syncthreads();

    // A fragments (global, f32 -> f16): lane reads K=[kb..kb+7] and [kb+16..]
    const int kb = k0 + khalf * 8;
    v16h a_re  = pack16(rowre + kb, rowre + kb + 16);
    v16h a_im  = pack16(rowim + kb, rowim + kb + 16);
    v16h a_mim = -a_im;

    // prefetch next K-step's A data (global_prefetch_b8)
    if (kb + 32 + 23 < L_SEQ) {
      __builtin_prefetch(rowre + kb + 32, 0, 0);
      __builtin_prefetch(rowim + kb + 32, 0, 0);
    }

#pragma unroll
    for (int nt = 0; nt < 3; ++nt) {
      // B fragments: lane -> column, K=[8h..8h+7] and [16+8h..] per lane-half
      const f16_t* cb = cosB + ((nt * 16 + (lane & 15)) * 32 + khalf * 8);
      const f16_t* sb = sinB + ((nt * 16 + (lane & 15)) * 32 + khalf * 8);
      v16h b_cos = cat8(*(const v8h*)cb, *(const v8h*)(cb + 16));
      v16h b_sin = cat8(*(const v8h*)sb, *(const v8h*)(sb + 16));
      // complex MAC: Re += re*cos - im*sin ; Im += re*sin + im*cos
      accRe[nt] = wmma_f16(a_re,  b_cos, accRe[nt]);
      accRe[nt] = wmma_f16(a_mim, b_sin, accRe[nt]);
      accIm[nt] = wmma_f16(a_re,  b_sin, accIm[nt]);
      accIm[nt] = wmma_f16(a_im,  b_cos, accIm[nt]);
    }
  }

  // |h|^2 and reduction over all 64 rx rows (each lane owns 8 rows per col)
  __syncthreads();
#pragma unroll
  for (int nt = 0; nt < 3; ++nt) {
    float p = 0.f;
#pragma unroll
    for (int j = 0; j < 8; ++j)
      p += accRe[nt][j] * accRe[nt][j] + accIm[nt][j] * accIm[nt][j];
    atomicAdd(&pow_s[nt * 16 + (lane & 15)], p);
  }
  __syncthreads();
  if (tid < 48) h_power[b * L_SEQ + tt + tid] = pow_s[tid];
}

// ---------------------------------------------------------------------------
// K2: per (b,port) windowed argmax of h_power -> m mod L and t_off mod L
// ---------------------------------------------------------------------------
__global__ __launch_bounds__(128)
void k2_argmax(const float* __restrict__ h_power,
               const int* __restrict__ cyclic_shifts,
               int* __restrict__ mmod, int* __restrict__ tmod) {
  const int tid = threadIdx.x;
  const int b = tid >> 4, p = tid & 15;
  const int cs   = cyclic_shifts[p];
  const int peak = ((8 - cs) % 8) * (L_SEQ / 8);
  const float* hp = h_power + b * L_SEQ;
  float best = -1.0f;      // powers are >= 0; strict '>' keeps first max
  int bestOff = -102;
  for (int off = -102; off < 102; ++off) {
    int idx = peak + off;
    if (idx < 0) idx += L_SEQ;
    if (idx >= L_SEQ) idx -= L_SEQ;
    float v = hp[idx];
    if (v > best) { best = v; bestOff = off; }
  }
  const int m  = bestOff + peak;
  mmod[b * NPORT + p] = ((m % L_SEQ) + L_SEQ) % L_SEQ;
  tmod[b * NPORT + p] = ((bestOff % L_SEQ) + L_SEQ) % L_SEQ;
}

// ---------------------------------------------------------------------------
// K3: one workgroup per (b,r). LDS-resident pipeline + WMMA 48x48 complex GEMM
// ---------------------------------------------------------------------------
__global__ __launch_bounds__(256)
void k3_main(const float* __restrict__ ls_re, const float* __restrict__ ls_im,
             const float* __restrict__ W_re,  const float* __restrict__ W_im,
             const int* __restrict__ mmod,    const int* __restrict__ tmod,
             float* __restrict__ out) {
  __shared__ float2 tab[L_SEQ];                 // e^{i 2pi q / L}
  __shared__ float lsr[L_SEQ];                  // ls row -> later residual (re)
  __shared__ float lsi[L_SEQ];                  // ls row -> later residual (im)
  __shared__ float havg[NPORT * NAVG * 2];      // interleaved re,im
  __shared__ f16_t hwr[272 * 64];               // 8 ports x 34 chunks, K pad 64
  __shared__ f16_t hwi[272 * 64];
  __shared__ f16_t wtr[48 * 64];                // W^T (d-major), K pad 64
  __shared__ f16_t wti[48 * 64];
  __shared__ int   msh[NPORT], tsh[NPORT];

  const int tid  = threadIdx.x;
  const int wave = tid >> 5;
  const int lane = tid & 31;
  const int r = blockIdx.x, b = blockIdx.y;

  if (tid < NPORT) { msh[tid] = mmod[b * NPORT + tid]; tsh[tid] = tmod[b * NPORT + tid]; }

  for (int q = tid; q < L_SEQ; q += 256) {       // phasor table (only sincos)
    float s, c; __sincosf(TWO_PI_OVER_L * (float)q, &s, &c);
    tab[q] = make_float2(c, s);
  }
  const size_t rowoff = ((size_t)b * NRX + r) * L_SEQ;
  for (int n = tid; n < L_SEQ; n += 256) {
    lsr[n] = ls_re[rowoff + n];
    lsi[n] = ls_im[rowoff + n];
  }
  for (int i = tid; i < 48 * 64; i += 256) {     // W^T, f16, zero-padded K
    int d = i >> 6, k = i & 63;
    float vr = 0.f, vi = 0.f;
    if (k < 48) { vr = W_re[k * 48 + d]; vi = W_im[k * 48 + d]; }
    wtr[i] = (f16_t)vr; wti[i] = (f16_t)vi;
  }
  for (int i = tid; i < 272 * 16; i += 256) {    // zero K-pad region of hwr/hwi
    int rowc = i >> 4, k = 48 + (i & 15);
    hwr[rowc * 64 + k] = (f16_t)0.f;
    hwi[rowc * 64 + k] = (f16_t)0.f;
  }
  __syncthreads();

  // Phase A: h_avg[p][g] = mean_{j<4} ls[4g+j] * e^{+i 2pi m (4g+j)/L}
  for (int i = tid; i < NPORT * NAVG; i += 256) {
    int p = i / NAVG, g = i % NAVG;
    int mm = msh[p];
    float ar = 0.f, ai = 0.f;
#pragma unroll
    for (int j = 0; j < 4; ++j) {
      int n = 4 * g + j;
      float2 e = tab[(mm * n) % L_SEQ];
      float xr = lsr[n], xi = lsi[n];
      ar += xr * e.x - xi * e.y;
      ai += xr * e.y + xi * e.x;
    }
    havg[i * 2 + 0] = ar * 0.25f;
    havg[i * 2 + 1] = ai * 0.25f;
  }
  __syncthreads();

  // Phase B: residual[n] = ls[n] - sum_p h_interp[p][n] * conj(phasor_m)
  for (int n = tid; n < L_SEQ; n += 256) {
    int kk = (n >= 2) ? ((n - 2) >> 2) : 0;
    if (kk > 406) kk = 406;
    float xc = (float)n - 1.5f;
    float recr = 0.f, reci = 0.f;
#pragma unroll
    for (int p = 0; p < NPORT; ++p) {
      const float* hb = havg + (p * NAVG + kk) * 2;
      float a_r = hb[0], a_i = hb[1];
      float sr = (hb[2] - a_r) * 0.25f;
      float si = (hb[3] - a_i) * 0.25f;
      float hr = sr * xc + a_r;
      float hi = si * xc + a_i;
      float2 e = tab[(msh[p] * n) % L_SEQ];
      recr += hr * e.x + hi * e.y;        // * conj(e^{i theta})
      reci += hi * e.x - hr * e.y;
    }
    lsr[n] -= recr;                        // lsr/lsi now hold the residual
    lsi[n] -= reci;
  }
  __syncthreads();

  // Phase C: two port-groups of 8 -> 272-row x 48 GEMM with W (complex)
  for (int pg = 0; pg < 2; ++pg) {
    // build h_with_resid in f16, chunk-major with 64-padded K
    for (int i = tid; i < 8 * L_SEQ; i += 256) {
      int pl = i / L_SEQ, n = i % L_SEQ;
      int p = pg * 8 + pl;
      int kk = (n >= 2) ? ((n - 2) >> 2) : 0;
      if (kk > 406) kk = 406;
      float xc = (float)n - 1.5f;
      const float* hb = havg + (p * NAVG + kk) * 2;
      float a_r = hb[0], a_i = hb[1];
      float sr = (hb[2] - a_r) * 0.25f;
      float si = (hb[3] - a_i) * 0.25f;
      float hr = sr * xc + a_r;
      float hi = si * xc + a_i;
      float2 e = tab[(msh[p] * n) % L_SEQ];
      float rr = lsr[n], ri = lsi[n];
      float wr_ = hr + rr * e.x - ri * e.y;
      float wi_ = hi + rr * e.y + ri * e.x;
      int ch = n / 48, k = n % 48;
      int idx = (pl * NCHUNK + ch) * 64 + k;
      hwr[idx] = (f16_t)wr_;
      hwi[idx] = (f16_t)wi_;
    }
    __syncthreads();

    // GEMM: rows = pl*34 + chunk (272), cols = d (48), K = 48 (padded 64)
    for (int mt = wave; mt < 17; mt += 8) {
      const int rowA  = mt * 16 + (lane & 15);
      const int khalf = lane >> 4;
      const f16_t* arb = hwr + rowA * 64;
      const f16_t* aib = hwi + rowA * 64;
      v8f accR[3] = {}, accI[3] = {};
#pragma unroll
      for (int k0 = 0; k0 < 64; k0 += 32) {
        const int kb = k0 + khalf * 8;
        v16h a_re  = cat8(*(const v8h*)(arb + kb), *(const v8h*)(arb + kb + 16));
        v16h a_im  = cat8(*(const v8h*)(aib + kb), *(const v8h*)(aib + kb + 16));
        v16h a_mim = -a_im;
#pragma unroll
        for (int nt = 0; nt < 3; ++nt) {
          const f16_t* br = wtr + ((nt * 16 + (lane & 15)) * 64 + k0 + khalf * 8);
          const f16_t* bi = wti + ((nt * 16 + (lane & 15)) * 64 + k0 + khalf * 8);
          v16h b_re = cat8(*(const v8h*)br, *(const v8h*)(br + 16));
          v16h b_im = cat8(*(const v8h*)bi, *(const v8h*)(bi + 16));
          accR[nt] = wmma_f16(a_re,  b_re, accR[nt]);
          accR[nt] = wmma_f16(a_mim, b_im, accR[nt]);
          accI[nt] = wmma_f16(a_re,  b_im, accI[nt]);
          accI[nt] = wmma_f16(a_im,  b_re, accI[nt]);
        }
      }
      // epilogue: * conj(phasor_T), scattered global float2 stores
      const int dl = lane & 15;
#pragma unroll
      for (int nt = 0; nt < 3; ++nt) {
#pragma unroll
        for (int j = 0; j < 8; ++j) {
          int mr  = (lane >> 4) * 8 + j;     // C/D layout: VGPR j -> row
          int row = mt * 16 + mr;
          int pl  = row / NCHUNK, ch = row % NCHUNK;
          int p   = pg * 8 + pl;
          int n   = ch * 48 + nt * 16 + dl;
          float2 e = tab[(tsh[p] * n) % L_SEQ];
          float re = accR[nt][j], im = accI[nt][j];
          float orr = re * e.x + im * e.y;   // * conj(phasor_T)
          float oii = im * e.x - re * e.y;
          size_t off = (((size_t)b * NRX + r) * NPORT + p) * L_SEQ + n;
          reinterpret_cast<float2*>(out)[off] = make_float2(orr, oii);
        }
      }
    }
    __syncthreads();
  }
}

// ---------------------------------------------------------------------------
extern "C" void kernel_launch(void* const* d_in, const int* in_sizes, int n_in,
                              void* d_out, int out_size, void* d_ws, size_t ws_size,
                              hipStream_t stream) {
  (void)in_sizes; (void)n_in; (void)out_size; (void)ws_size;
  const float* ls_re = (const float*)d_in[0];
  const float* ls_im = (const float*)d_in[1];
  const int*   cyc   = (const int*)  d_in[2];
  const float* W_re  = (const float*)d_in[3];
  const float* W_im  = (const float*)d_in[4];

  float* hpow = (float*)d_ws;                          // 8*1632 f32
  int*   mmod = (int*)((char*)d_ws + 8 * L_SEQ * 4);   // 128 int
  int*   tmod = mmod + 8 * NPORT;                      // 128 int

  k1_dft_power<<<dim3(L_SEQ / 48, 8), 128, 0, stream>>>(ls_re, ls_im, hpow);
  k2_argmax  <<<1, 128, 0, stream>>>(hpow, cyc, mmod, tmod);
  k3_main    <<<dim3(NRX, 8), 256, 0, stream>>>(ls_re, ls_im, W_re, W_im,
                                                mmod, tmod, (float*)d_out);
}
